// MrLoraLayer_86208583566010
// MI455X (gfx1250) — compile-verified, matrix operand order checked by reading, not run on hardware
//
#include <hip/hip_runtime.h>
#include <hip/hip_bf16.h>

// ---------------------------------------------------------------------------
// MrLoRA fused layer for gfx1250 (MI455X), compile-only target.
//
// out[m,n] = sum_k x[m,k]*W[n,k] + bias[n] + sum_r z[m,r]*Bpad[n,r]
// with z = (x @ A^T) * combined (padded 63 -> 64), Bpad = B padded 63 -> 64.
//
// Main GEMM: bf16x3 split-fp32 via v_wmma_f32_16x16x32_bf16.
// LoRA adapter folded in as 2 extra K-chunks of the same loop.
// LDS tiles use a 16B-granule XOR swizzle: conflict-free ds_load_b128
// fragment reads and ds_store_b64 stores while keeping LDS at exactly 64KB.
// ---------------------------------------------------------------------------

typedef __attribute__((ext_vector_type(16))) __bf16 v16bf;
typedef __attribute__((ext_vector_type(8)))  __bf16 v8bf;
typedef __attribute__((ext_vector_type(8)))  float  v8f;
typedef __attribute__((ext_vector_type(4)))  unsigned short us4;

#define IN_F   4096
#define OUT_F  4096
#define MTOT   8192          // 4 * 2048
#define RPAD   64            // 63 LoRA ranks padded to 64
#define BM     128
#define BN     128
#define KC     32            // bf16 K per WMMA chunk
#define NKT    ((IN_F / KC) + (RPAD / KC))   // 128 + 2 = 130 chunks
#define LDSTR  32            // ushorts per LDS tile row (32 bf16 = 64B)
#define TILE_US (BM * LDSTR) // 4096 ushorts = 8KB per tile
// LDS: 2 stages * 4 tiles (Ahi, Alo, Whi, Wlo) * 8KB = 64KB dynamic

// XOR swizzle (in ushort units): physical = row*LDSTR + (granule8 ^ swz8) + sub
// granule8 = (logical 16B granule)*8 ushorts, swz8 = ((row>>2)&3)*8.
__device__ inline int swz_off(int row, int granule8, int sub) {
    return row * LDSTR + ((granule8 ^ (((row >> 2) & 3) << 3))) + sub;
}

// ---------------------------------------------------------------------------
// Kernel 0: pad LoRA B [OUT_F, 63] -> Bpad [OUT_F, 64] (col 63 = 0)
// ---------------------------------------------------------------------------
__global__ void __launch_bounds__(256) bpad_kernel(const float* __restrict__ Bsrc,
                                                   float* __restrict__ Bpad) {
    int idx = blockIdx.x * 256 + threadIdx.x;   // OUT_F*64 threads
    int n = idx >> 6;
    int j = idx & 63;
    Bpad[idx] = (j < 63) ? Bsrc[(size_t)n * 63 + j] : 0.0f;
}

// ---------------------------------------------------------------------------
// Kernel 1: z[m, j] = comb[j] * sum_k x[m,k] * A[j,k], j padded to 64 (j=63 -> 0)
// RANKS = [32,16,8,4,2,1]; segment id from j; comb = alphas[s]*scalings[s].
// Block: 256 threads handle a 64(m) x 64(j) tile; K chunked by 64 via LDS.
// ---------------------------------------------------------------------------
__global__ void __launch_bounds__(256) zproj_kernel(const float* __restrict__ x,
                                                    const float* __restrict__ A,
                                                    const float* __restrict__ alphas,
                                                    const float* __restrict__ scalings,
                                                    float* __restrict__ z) {
    __shared__ float xs[64 * 64];
    __shared__ float as[64 * 65];   // +1 pad: conflict-free lane-varied j reads
    const int t  = threadIdx.x;
    const int m0 = blockIdx.x * 64;
    const int jj   = t & 63;        // rank column (lane-varied within a wave)
    const int quad = t >> 6;        // 0..3 (constant within a wave -> broadcast)

    float acc[16];
#pragma unroll
    for (int r = 0; r < 16; ++r) acc[r] = 0.0f;

    for (int kc = 0; kc < IN_F; kc += 64) {
#pragma unroll
        for (int i = 0; i < 16; ++i) {
            int lin = t + 256 * i;          // 0..4095
            int row = lin >> 6;
            int col = lin & 63;
            xs[row * 64 + col] = x[(size_t)(m0 + row) * IN_F + kc + col];
            as[row * 65 + col] = (row < 63) ? A[(size_t)row * IN_F + kc + col] : 0.0f;
        }
        __syncthreads();
#pragma unroll 4
        for (int rr = 0; rr < 16; ++rr) {
            int row = quad + 4 * rr;
            float s = acc[rr];
#pragma unroll
            for (int k = 0; k < 64; ++k)
                s = fmaf(xs[row * 64 + k], as[jj * 65 + k], s);
            acc[rr] = s;
        }
        __syncthreads();
    }

    int sg = (jj < 32) ? 0 : (jj < 48) ? 1 : (jj < 56) ? 2 : (jj < 60) ? 3 : (jj < 62) ? 4 : 5;
    float comb = (jj < 63) ? alphas[sg] * scalings[sg] : 0.0f;
#pragma unroll
    for (int rr = 0; rr < 16; ++rr)
        z[(size_t)(m0 + quad + 4 * rr) * RPAD + jj] = acc[rr] * comb;
}

// ---------------------------------------------------------------------------
// Fragment loader: 16-bit A-matrix 16x32 wave32 layout (ISA 7.12.2).
// Lane L<16 holds row M=L, K in {0..7} then {16..23}; L>=16: row M=L-16,
// K in {8..15} then {24..31}. Logical 16B granules per row: g0=K0-7, g1=K8-15,
// g2=K16-23, g3=K24-31; lane<16 reads g{half}, g{half+2} with half=lane>>4.
// Physical placement is XOR-swizzled; (l+2)*8 == (l|2)*8 for l in {0,1}, so
// the +16-ushort step commutes with the XOR.
// ---------------------------------------------------------------------------
__device__ inline v16bf load_frag(const unsigned short* tile, int row0, int lane) {
    const int half = (lane >> 4) & 1;
    const int row  = row0 + (lane & 15);
    const int g0   = half << 3;                         // logical granule*8
    const unsigned short* p0 = tile + swz_off(row, g0, 0);
    const unsigned short* p1 = tile + swz_off(row, g0 + 16, 0);
    v8bf a = *(const v8bf*)p0;        // K 0..7  (or 8..15)
    v8bf b = *(const v8bf*)p1;        // K 16..23 (or 24..31)
    return __builtin_shufflevector(a, b, 0, 1, 2, 3, 4, 5, 6, 7,
                                         8, 9, 10, 11, 12, 13, 14, 15);
}

// ---------------------------------------------------------------------------
// Main fused GEMM. Block tile 128x128, 8 waves, wave tile 64x32 (4x2 WMMA
// tiles). bf16x3: acc += xhi*whi + xhi*wlo + xlo*whi, fp32 accumulate.
// ---------------------------------------------------------------------------
__global__ void __launch_bounds__(256) mrlora_gemm_kernel(
    const float* __restrict__ X, const float* __restrict__ W,
    const float* __restrict__ bias, const float* __restrict__ Z,
    const float* __restrict__ Bp, float* __restrict__ out) {
    extern __shared__ unsigned short smem[];   // [2][4][TILE_US] = 64KB

    const int t    = threadIdx.x;
    const int lane = t & 31;
    const int wave = t >> 5;      // 0..7
    const int wm   = wave >> 2;   // 0..1 -> 64-row slab
    const int wn   = wave & 3;    // 0..3 -> 32-col slab
    const int m0   = blockIdx.y * BM;
    const int n0   = blockIdx.x * BN;

    const int c  = (t & 7) * 4;   // K column (floats / bf16 elems), 0..28 step 4
    const int r0 = t >> 3;        // 0..31 row base (4 rows strided by 32)
    const int cg8  = (c >> 3) << 3;  // logical 16B granule * 8 ushorts
    const int csub = c & 7;          // 0 or 4: 8B half within the granule

    const v8f vzero = {0.f, 0.f, 0.f, 0.f, 0.f, 0.f, 0.f, 0.f};
    v8f acc[4][2];
#pragma unroll
    for (int i = 0; i < 4; ++i)
#pragma unroll
        for (int j = 0; j < 2; ++j) acc[i][j] = vzero;

    float4 ra[4], rw[4];   // register staging for the next K chunk

    auto loadRegs = [&](int kt) {
        const float *As, *Ws;
        int lda, ldw;
        if (kt < IN_F / KC) {
            As = X + (size_t)m0 * IN_F + kt * KC;  lda = IN_F;
            Ws = W + (size_t)n0 * IN_F + kt * KC;  ldw = IN_F;
        } else {
            int k2 = (kt - IN_F / KC) * KC;        // adapter chunks: z @ Bpad^T
            As = Z  + (size_t)m0 * RPAD + k2;      lda = RPAD;
            Ws = Bp + (size_t)n0 * RPAD + k2;      ldw = RPAD;
        }
#pragma unroll
        for (int i = 0; i < 4; ++i) {
            ra[i] = *(const float4*)(As + (size_t)(r0 + 32 * i) * lda + c);
            rw[i] = *(const float4*)(Ws + (size_t)(r0 + 32 * i) * ldw + c);
        }
    };

    auto storeLDS = [&](int stage) {
        unsigned short* base = smem + (size_t)stage * 4 * TILE_US;
        unsigned short* ahi = base;
        unsigned short* alo = base + TILE_US;
        unsigned short* whi = base + 2 * TILE_US;
        unsigned short* wlo = base + 3 * TILE_US;
#pragma unroll
        for (int i = 0; i < 4; ++i) {
            int row = r0 + 32 * i;
            int off = swz_off(row, cg8, csub);
            float fa[4] = {ra[i].x, ra[i].y, ra[i].z, ra[i].w};
            float fw[4] = {rw[i].x, rw[i].y, rw[i].z, rw[i].w};
            us4 ha, la, hw, lw;
#pragma unroll
            for (int e = 0; e < 4; ++e) {
                unsigned ua  = __float_as_uint(fa[e]);
                unsigned uah = ua & 0xFFFF0000u;
                float    fal = fa[e] - __uint_as_float(uah);
                ha[e] = (unsigned short)(uah >> 16);
                la[e] = (unsigned short)(__float_as_uint(fal) >> 16);
                unsigned uw  = __float_as_uint(fw[e]);
                unsigned uwh = uw & 0xFFFF0000u;
                float    fwl = fw[e] - __uint_as_float(uwh);
                hw[e] = (unsigned short)(uwh >> 16);
                lw[e] = (unsigned short)(__float_as_uint(fwl) >> 16);
            }
            *(us4*)(ahi + off) = ha;
            *(us4*)(alo + off) = la;
            *(us4*)(whi + off) = hw;
            *(us4*)(wlo + off) = lw;
        }
    };

    auto compute = [&](int stage) {
        const unsigned short* base = smem + (size_t)stage * 4 * TILE_US;
        const unsigned short* ahi = base;
        const unsigned short* alo = base + TILE_US;
        const unsigned short* whi = base + 2 * TILE_US;
        const unsigned short* wlo = base + 3 * TILE_US;

        v16bf Ah[4], Bh[2];
#pragma unroll
        for (int i = 0; i < 4; ++i) Ah[i] = load_frag(ahi, wm * 64 + i * 16, lane);
#pragma unroll
        for (int j = 0; j < 2; ++j) Bh[j] = load_frag(whi, wn * 32 + j * 16, lane);
#pragma unroll
        for (int i = 0; i < 4; ++i)
#pragma unroll
            for (int j = 0; j < 2; ++j)
                acc[i][j] = __builtin_amdgcn_wmma_f32_16x16x32_bf16(
                    false, Ah[i], false, Bh[j], (short)0, acc[i][j], false, false);

        v16bf Bl[2];
#pragma unroll
        for (int j = 0; j < 2; ++j) Bl[j] = load_frag(wlo, wn * 32 + j * 16, lane);
#pragma unroll
        for (int i = 0; i < 4; ++i)
#pragma unroll
            for (int j = 0; j < 2; ++j)
                acc[i][j] = __builtin_amdgcn_wmma_f32_16x16x32_bf16(
                    false, Ah[i], false, Bl[j], (short)0, acc[i][j], false, false);

        v16bf Al[4];
#pragma unroll
        for (int i = 0; i < 4; ++i) Al[i] = load_frag(alo, wm * 64 + i * 16, lane);
#pragma unroll
        for (int i = 0; i < 4; ++i)
#pragma unroll
            for (int j = 0; j < 2; ++j)
                acc[i][j] = __builtin_amdgcn_wmma_f32_16x16x32_bf16(
                    false, Al[i], false, Bh[j], (short)0, acc[i][j], false, false);
    };

    // Software-pipelined main loop: 128 chunks of x/W + 2 adapter chunks.
    loadRegs(0);
    storeLDS(0);
    __syncthreads();
    for (int kt = 0; kt < NKT; ++kt) {
        if (kt + 1 < NKT) loadRegs(kt + 1);
        compute(kt & 1);
        if (kt + 1 < NKT) {
            storeLDS((kt + 1) & 1);
            __syncthreads();
        }
    }

    // Epilogue: bias add + store. C/D layout (ISA 7.12.2): lane L -> N = L%16,
    // VGPR v -> M = v + 8*(L/16).
    float bb[2];
#pragma unroll
    for (int j = 0; j < 2; ++j)
        bb[j] = bias[n0 + wn * 32 + j * 16 + (lane & 15)];
#pragma unroll
    for (int i = 0; i < 4; ++i) {
        int gmb = m0 + wm * 64 + i * 16 + ((lane >> 4) & 1) * 8;
#pragma unroll
        for (int j = 0; j < 2; ++j) {
            int gn = n0 + wn * 32 + j * 16 + (lane & 15);
#pragma unroll
            for (int v = 0; v < 8; ++v)
                out[(size_t)(gmb + v) * OUT_F + gn] = acc[i][j][v] + bb[j];
        }
    }
}

// ---------------------------------------------------------------------------
// Host entry. Inputs (setup_inputs order): x, W_base, b_base, A, B, alphas,
// scalings. Workspace: z [MTOT*RPAD] f32 then Bpad [OUT_F*RPAD] f32 (~3 MB).
// ---------------------------------------------------------------------------
extern "C" void kernel_launch(void* const* d_in, const int* in_sizes, int n_in,
                              void* d_out, int out_size, void* d_ws, size_t ws_size,
                              hipStream_t stream) {
    const float* x        = (const float*)d_in[0];
    const float* W_base   = (const float*)d_in[1];
    const float* b_base   = (const float*)d_in[2];
    const float* A        = (const float*)d_in[3];
    const float* B        = (const float*)d_in[4];
    const float* alphas   = (const float*)d_in[5];
    const float* scalings = (const float*)d_in[6];
    float* out  = (float*)d_out;

    float* z    = (float*)d_ws;                    // MTOT * RPAD
    float* Bpad = z + (size_t)MTOT * RPAD;         // OUT_F * RPAD

    bpad_kernel<<<(OUT_F * RPAD) / 256, 256, 0, stream>>>(B, Bpad);
    zproj_kernel<<<MTOT / 64, 256, 0, stream>>>(x, A, alphas, scalings, z);

    dim3 grid(OUT_F / BN, MTOT / BM);              // (32, 64)
    size_t ldsBytes = (size_t)2 * 4 * TILE_US * sizeof(unsigned short);  // 65536
    mrlora_gemm_kernel<<<grid, 256, ldsBytes, stream>>>(x, W_base, b_base, z, Bpad, out);
}
